// MultiHeadAttention_16243566313764
// MI455X (gfx1250) — matrix operand order repaired and checked by reading, not compiled, not run on hardware
//
#include <hip/hip_runtime.h>

typedef __attribute__((ext_vector_type(16))) _Float16 v16h;
typedef __attribute__((ext_vector_type(8)))  _Float16 v8h;
typedef __attribute__((ext_vector_type(8)))  float    v8f;

#define D_MODEL 1024
#define NHEAD   16
#define HEAD_DIM 64
#define SEQ     2048
#define BATCH   4

__device__ __forceinline__ v8f wmma_f32_f16(v16h a, v16h b, v8f c) {
  // D = A(16x32 f16) * B(32x16 f16) + C(16x16 f32)
  return __builtin_amdgcn_wmma_f32_16x16x32_f16(false, a, false, b, (short)0, c,
                                                false, false);
}

// A fragment: 16x32 f16 tile, p0 -> element (row0, k0) of row-major LDS tile.
// Lanes 0-15 = rows with K 0-15; lanes 16-31 same rows, K 16-31.
__device__ __forceinline__ v16h lds_load_a(const _Float16* p0, int ld) {
  const int lane = threadIdx.x & 31;
  const _Float16* p = p0 + (lane & 15) * ld + ((lane >> 4) << 4);
  v8h lo = *(const v8h*)p;
  v8h hi = *(const v8h*)(p + 8);
  v16h a;
#pragma unroll
  for (int i = 0; i < 8; ++i) { a[i] = lo[i]; a[i + 8] = hi[i]; }
  return a;
}

// B fragment: 32x16 where B(k,n) = src[n][k], src row-major [N][K].
// Lane n holds 16 consecutive k (lanes 0-15: K 0-15, lanes 16-31: K 16-31).
__device__ __forceinline__ v16h lds_load_bt(const _Float16* p0, int ld) {
  const int lane = threadIdx.x & 31;
  const _Float16* p = p0 + (lane & 15) * ld + ((lane >> 4) << 4);
  v8h lo = *(const v8h*)p;
  v8h hi = *(const v8h*)(p + 8);
  v16h b;
#pragma unroll
  for (int i = 0; i < 8; ++i) { b[i] = lo[i]; b[i + 8] = hi[i]; }
  return b;
}

// B fragment: 32x16 where B(k,n) = src[k][n], src row-major [K][N] (strided).
__device__ __forceinline__ v16h lds_load_b(const _Float16* p0, int ld) {
  const int lane = threadIdx.x & 31;
  const _Float16* p = p0 + ((lane >> 4) << 4) * ld + (lane & 15);
  v16h b;
#pragma unroll
  for (int i = 0; i < 16; ++i) b[i] = p[i * ld];
  return b;
}

// Async raw copy: 16B global -> LDS, tracked by ASYNCcnt (CDNA5 VGLOBAL op 98).
__device__ __forceinline__ void async_b128(unsigned int lds_off,
                                           const _Float16* gptr) {
  asm volatile("global_load_async_to_lds_b128 %0, %1, off"
               :: "v"(lds_off), "v"(gptr) : "memory");
}

// ---------------------------------------------------------------------------
// Kernel 1: fused Q/K/V projection: out = X @ W^T + bias (fp32 in, f16 out in
// [B,H,L,Hd]).  fp32->f16 conversion folded into LDS staging; register-
// prefetch pipeline overlaps next tile's global loads with WMMA compute.
// ---------------------------------------------------------------------------
__global__ __launch_bounds__(256) void qkv_proj_kernel(
    const float* __restrict__ xq, const float* __restrict__ xk,
    const float* __restrict__ xv, const float* __restrict__ wq,
    const float* __restrict__ wk, const float* __restrict__ wv,
    const float* __restrict__ bq, const float* __restrict__ bk,
    const float* __restrict__ bv, _Float16* __restrict__ Qo,
    _Float16* __restrict__ Ko, _Float16* __restrict__ Vo) {
  __shared__ _Float16 As[128][40];
  __shared__ _Float16 Bs[128][40];

  const int z = blockIdx.z;
  const float* X    = (z == 0) ? xq : (z == 1) ? xk : xv;
  const float* W    = (z == 0) ? wq : (z == 1) ? wk : wv;
  const float* Bias = (z == 0) ? bq : (z == 1) ? bk : bv;
  _Float16*    Out  = (z == 0) ? Qo : (z == 1) ? Ko : Vo;

  const int m0 = blockIdx.x * 128;
  const int n0 = blockIdx.y * 128;
  const int tid = threadIdx.x;
  const int lane = tid & 31;
  const int wid = tid >> 5;
  const int wm = (wid >> 1) * 32;
  const int wn = (wid & 1) * 64;

  v8f acc[2][4] = {};
  float4 xr[4], wr[4];

  // prologue: fetch tile k0=0 into registers
#pragma unroll
  for (int j = 0; j < 4; ++j) {
    int fid = j * 256 + tid;
    int row = fid >> 3;
    int kc  = (fid & 7) << 2;
    xr[j] = *(const float4*)(X + (size_t)(m0 + row) * D_MODEL + kc);
    wr[j] = *(const float4*)(W + (size_t)(n0 + row) * D_MODEL + kc);
  }

  for (int k0 = 0; k0 < D_MODEL; k0 += 32) {
    __syncthreads();
#pragma unroll
    for (int j = 0; j < 4; ++j) {
      int fid = j * 256 + tid;
      int row = fid >> 3;
      int kc  = (fid & 7) << 2;
      As[row][kc + 0] = (_Float16)xr[j].x; As[row][kc + 1] = (_Float16)xr[j].y;
      As[row][kc + 2] = (_Float16)xr[j].z; As[row][kc + 3] = (_Float16)xr[j].w;
      Bs[row][kc + 0] = (_Float16)wr[j].x; Bs[row][kc + 1] = (_Float16)wr[j].y;
      Bs[row][kc + 2] = (_Float16)wr[j].z; Bs[row][kc + 3] = (_Float16)wr[j].w;
    }
    __syncthreads();

    if (k0 + 32 < D_MODEL) {  // fetch next tile while WMMAs run
#pragma unroll
      for (int j = 0; j < 4; ++j) {
        int fid = j * 256 + tid;
        int row = fid >> 3;
        int kc  = (fid & 7) << 2;
        xr[j] = *(const float4*)(X + (size_t)(m0 + row) * D_MODEL + k0 + 32 + kc);
        wr[j] = *(const float4*)(W + (size_t)(n0 + row) * D_MODEL + k0 + 32 + kc);
      }
    }

    v16h a0 = lds_load_a(&As[wm][0], 40);
    v16h a1 = lds_load_a(&As[wm + 16][0], 40);
#pragma unroll
    for (int nf = 0; nf < 4; ++nf) {
      v16h bb = lds_load_bt(&Bs[wn + nf * 16][0], 40);
      acc[0][nf] = wmma_f32_f16(a0, bb, acc[0][nf]);
      acc[1][nf] = wmma_f32_f16(a1, bb, acc[1][nf]);
    }
  }

#pragma unroll
  for (int nf = 0; nf < 4; ++nf) {
    int n = n0 + wn + nf * 16 + (lane & 15);
    float bias = Bias[n];
    int h = n >> 6, hd = n & 63;
#pragma unroll
    for (int mf = 0; mf < 2; ++mf) {
#pragma unroll
      for (int v = 0; v < 8; ++v) {
        int m = m0 + wm + mf * 16 + ((lane >> 4) << 3) + v;
        int bb = m >> 11, l = m & (SEQ - 1);
        size_t idx = (((((size_t)bb * NHEAD + h) << 11) | l) << 6) | hd;
        Out[idx] = (_Float16)(acc[mf][nf][v] + bias);
      }
    }
  }
}

// ---------------------------------------------------------------------------
// Kernel 2: flash attention per (b,h).  Q tile register-resident per wave;
// K/V 64-key tiles double-buffered via global_load_async_to_lds_b128
// (ASYNCcnt) overlapped with WMMA compute + online softmax.
// ---------------------------------------------------------------------------
__global__ __launch_bounds__(256) void attn_kernel(
    const _Float16* __restrict__ Q, const _Float16* __restrict__ K,
    const _Float16* __restrict__ V, _Float16* __restrict__ CTX) {
  // Pool: [0,18432) Ks[2][64][72] (overlaid on Qs[128][72], dead after frag
  // load); [18432,36864) Vs[2][64][72]; [36864,55296) Ps[8][16][72].
  __shared__ __align__(16) unsigned char smem[55296];
  _Float16 (*Qs)[72]     = (_Float16(*)[72])smem;
  _Float16 (*Ks)[64][72] = (_Float16(*)[64][72])smem;
  _Float16 (*Vs)[64][72] = (_Float16(*)[64][72])(smem + 18432);
  _Float16 (*Ps)[16][72] = (_Float16(*)[16][72])(smem + 36864);

  const int bh = blockIdx.y;
  const int b = bh >> 4;
  const int h = bh & 15;
  const int q0 = blockIdx.x * 128;
  const int tid = threadIdx.x;
  const int lane = tid & 31;
  const int wid = tid >> 5;
  const int lhi = lane >> 4;
  const int llo = lane & 15;

  const _Float16* Qb = Q + (size_t)bh * SEQ * HEAD_DIM;
  const _Float16* Kb = K + (size_t)bh * SEQ * HEAD_DIM;
  const _Float16* Vb = V + (size_t)bh * SEQ * HEAD_DIM;

  // Stage Q tile (128 x 64 halves) once, then move to registers.
#pragma unroll
  for (int j = 0; j < 4; ++j) {
    int fid = j * 256 + tid;
    int row = fid >> 3;
    int c = (fid & 7) << 3;
    *(v8h*)&Qs[row][c] = *(const v8h*)(Qb + (size_t)(q0 + row) * HEAD_DIM + c);
  }
  __syncthreads();

  const int qrow = wid * 16;
  v16h aq0 = lds_load_a(&Qs[qrow][0], 72);
  v16h aq1 = lds_load_a(&Qs[qrow][32], 72);
#pragma unroll
  for (int i = 0; i < 16; ++i) {  // fold 1/sqrt(Hd)=0.125 into Q
    aq0[i] = (_Float16)((float)aq0[i] * 0.125f);
    aq1[i] = (_Float16)((float)aq1[i] * 0.125f);
  }
  __syncthreads();  // everyone done with Qs before async K writes land there

  // Issue async copy of K/V tile 0 into buffer 0 (4 ops per wave).
#pragma unroll
  for (int j = 0; j < 2; ++j) {
    int fid = j * 256 + tid;
    int row = fid >> 3;
    int c = (fid & 7) << 3;
    async_b128((unsigned int)(unsigned long long)&Ks[0][row][c],
               Kb + (size_t)row * HEAD_DIM + c);
    async_b128((unsigned int)(unsigned long long)&Vs[0][row][c],
               Vb + (size_t)row * HEAD_DIM + c);
  }

  float mrun[8], lrun[8];
  v8f o[4] = {};
#pragma unroll
  for (int v = 0; v < 8; ++v) { mrun[v] = -1e30f; lrun[v] = 0.f; }

  const int NT = SEQ / 64;
  for (int kt = 0; kt < NT; ++kt) {
    const int cur = kt & 1;
    __syncthreads();  // all waves done reading buffer (kt+1)&1 from iter kt-1
    if (kt + 1 < NT) {
      const int nxt = (kt + 1) & 1;
#pragma unroll
      for (int j = 0; j < 2; ++j) {
        int fid = j * 256 + tid;
        int row = fid >> 3;
        int c = (fid & 7) << 3;
        async_b128((unsigned int)(unsigned long long)&Ks[nxt][row][c],
                   Kb + (size_t)((kt + 1) * 64 + row) * HEAD_DIM + c);
        async_b128((unsigned int)(unsigned long long)&Vs[nxt][row][c],
                   Vb + (size_t)((kt + 1) * 64 + row) * HEAD_DIM + c);
      }
      asm volatile("s_wait_asynccnt 0x4" ::: "memory");  // tile kt landed
    } else {
      asm volatile("s_wait_asynccnt 0x0" ::: "memory");
    }
    __syncthreads();

    // S = (Q/8) @ K^T : 16 q x 64 keys per wave
    v8f s[4] = {};
#pragma unroll
    for (int nf = 0; nf < 4; ++nf) {
      v16h bk0 = lds_load_bt(&Ks[cur][nf * 16][0], 72);
      v16h bk1 = lds_load_bt(&Ks[cur][nf * 16][32], 72);
      s[nf] = wmma_f32_f16(aq0, bk0, s[nf]);
      s[nf] = wmma_f32_f16(aq1, bk1, s[nf]);
    }

    // Online softmax.  D-frag: row = lhi*8+v, col = llo + 16*nf; row
    // reductions are xor-shuffles across the 16 lanes of each half-wave.
#pragma unroll
    for (int v = 0; v < 8; ++v) {
      float pm = fmaxf(fmaxf(s[0][v], s[1][v]), fmaxf(s[2][v], s[3][v]));
#pragma unroll
      for (int off = 8; off >= 1; off >>= 1) pm = fmaxf(pm, __shfl_xor(pm, off, 32));
      float mnew = fmaxf(mrun[v], pm);
      float corr = __expf(mrun[v] - mnew);
      mrun[v] = mnew;
      float rs = 0.f;
      int prow = (lhi << 3) + v;
#pragma unroll
      for (int nf = 0; nf < 4; ++nf) {
        float pv = __expf(s[nf][v] - mnew);
        rs += pv;
        Ps[wid][prow][llo + (nf << 4)] = (_Float16)pv;  // D-layout -> LDS
      }
#pragma unroll
      for (int off = 8; off >= 1; off >>= 1) rs += __shfl_xor(rs, off, 32);
      lrun[v] = lrun[v] * corr + rs;
#pragma unroll
      for (int nf = 0; nf < 4; ++nf) o[nf][v] *= corr;
    }

    // Wave-private LDS round-trip (LDS completes in order within a wave).
    asm volatile("" ::: "memory");

    // O += P @ V   (V row-major in LDS -> strided B-frag loads)
    v16h ap0 = lds_load_a(&Ps[wid][0][0], 72);
    v16h ap1 = lds_load_a(&Ps[wid][0][32], 72);
#pragma unroll
    for (int nf = 0; nf < 4; ++nf) {
      v16h bv0 = lds_load_b(&Vs[cur][0][nf * 16], 72);
      v16h bv1 = lds_load_b(&Vs[cur][32][nf * 16], 72);
      o[nf] = wmma_f32_f16(ap0, bv0, o[nf]);
      o[nf] = wmma_f32_f16(ap1, bv1, o[nf]);
    }
  }

  // Normalize and store ctx as f16 [B, L, D] (col = h*64 + hd).
  float inv[8];
#pragma unroll
  for (int v = 0; v < 8; ++v) inv[v] = 1.0f / lrun[v];
#pragma unroll
  for (int nf = 0; nf < 4; ++nf) {
    int col = h * HEAD_DIM + (nf << 4) + llo;
#pragma unroll
    for (int v = 0; v < 8; ++v) {
      int l = q0 + qrow + (lhi << 3) + v;
      CTX[((size_t)b * SEQ + l) * D_MODEL + col] = (_Float16)(o[nf][v] * inv[v]);
    }
  }
}

// ---------------------------------------------------------------------------
// Kernel 3: output projection: out = ctx(f16) @ Wo^T + bo, fp32 result.
// Register-prefetch pipeline as in kernel 1.
// ---------------------------------------------------------------------------
__global__ __launch_bounds__(256) void out_proj_kernel(
    const _Float16* __restrict__ CTX, const float* __restrict__ wo,
    const float* __restrict__ bo, float* __restrict__ out) {
  __shared__ _Float16 As[128][40];
  __shared__ _Float16 Bs[128][40];

  const int m0 = blockIdx.x * 128;
  const int n0 = blockIdx.y * 128;
  const int tid = threadIdx.x;
  const int lane = tid & 31;
  const int wid = tid >> 5;
  const int wm = (wid >> 1) * 32;
  const int wn = (wid & 1) * 64;

  v8f acc[2][4] = {};
  v8h ar[2];
  float4 wr[4];

#pragma unroll
  for (int j = 0; j < 2; ++j) {
    int fid = j * 256 + tid;
    ar[j] = *(const v8h*)(CTX + (size_t)(m0 + (fid >> 2)) * D_MODEL +
                          ((fid & 3) << 3));
  }
#pragma unroll
  for (int j = 0; j < 4; ++j) {
    int fid = j * 256 + tid;
    wr[j] = *(const float4*)(wo + (size_t)(n0 + (fid >> 3)) * D_MODEL +
                             ((fid & 7) << 2));
  }

  for (int k0 = 0; k0 < D_MODEL; k0 += 32) {
    __syncthreads();
#pragma unroll
    for (int j = 0; j < 2; ++j) {
      int fid = j * 256 + tid;
      *(v8h*)&As[fid >> 2][(fid & 3) << 3] = ar[j];
    }
#pragma unroll
    for (int j = 0; j < 4; ++j) {
      int fid = j * 256 + tid;
      int row = fid >> 3;
      int kc = (fid & 7) << 2;
      Bs[row][kc + 0] = (_Float16)wr[j].x; Bs[row][kc + 1] = (_Float16)wr[j].y;
      Bs[row][kc + 2] = (_Float16)wr[j].z; Bs[row][kc + 3] = (_Float16)wr[j].w;
    }
    __syncthreads();

    if (k0 + 32 < D_MODEL) {
#pragma unroll
      for (int j = 0; j < 2; ++j) {
        int fid = j * 256 + tid;
        ar[j] = *(const v8h*)(CTX + (size_t)(m0 + (fid >> 2)) * D_MODEL +
                              k0 + 32 + ((fid & 3) << 3));
      }
#pragma unroll
      for (int j = 0; j < 4; ++j) {
        int fid = j * 256 + tid;
        wr[j] = *(const float4*)(wo + (size_t)(n0 + (fid >> 3)) * D_MODEL +
                                 k0 + 32 + ((fid & 7) << 2));
      }
    }

    v16h a0 = lds_load_a(&As[wm][0], 40);
    v16h a1 = lds_load_a(&As[wm + 16][0], 40);
#pragma unroll
    for (int nf = 0; nf < 4; ++nf) {
      v16h bb = lds_load_bt(&Bs[wn + nf * 16][0], 40);
      acc[0][nf] = wmma_f32_f16(a0, bb, acc[0][nf]);
      acc[1][nf] = wmma_f32_f16(a1, bb, acc[1][nf]);
    }
  }

#pragma unroll
  for (int nf = 0; nf < 4; ++nf) {
    int n = n0 + wn + nf * 16 + (lane & 15);
    float bias = bo[n];
#pragma unroll
    for (int mf = 0; mf < 2; ++mf) {
#pragma unroll
      for (int v = 0; v < 8; ++v) {
        int m = m0 + wm + mf * 16 + ((lane >> 4) << 3) + v;
        out[(size_t)m * D_MODEL + n] = acc[mf][nf][v] + bias;
      }
    }
  }
}

extern "C" void kernel_launch(void* const* d_in, const int* in_sizes, int n_in,
                              void* d_out, int out_size, void* d_ws, size_t ws_size,
                              hipStream_t stream) {
  (void)in_sizes; (void)n_in; (void)out_size; (void)ws_size;
  const float* query = (const float*)d_in[0];
  const float* key   = (const float*)d_in[1];
  const float* value = (const float*)d_in[2];
  const float* wq = (const float*)d_in[3];
  const float* bq = (const float*)d_in[4];
  const float* wk = (const float*)d_in[5];
  const float* bk = (const float*)d_in[6];
  const float* wv = (const float*)d_in[7];
  const float* bv = (const float*)d_in[8];
  const float* wo = (const float*)d_in[9];
  const float* bo = (const float*)d_in[10];
  float* out = (float*)d_out;

  const size_t nElem = (size_t)BATCH * SEQ * D_MODEL;  // 8,388,608
  _Float16* Q16 = (_Float16*)d_ws;       // [B,H,L,Hd] f16, 16 MiB
  _Float16* K16 = Q16 + nElem;           // 16 MiB
  _Float16* V16 = K16 + nElem;           // 16 MiB
  _Float16* CTX = V16 + nElem;           // [B,L,D] f16, 16 MiB

  qkv_proj_kernel<<<dim3(64, 8, 3), 256, 0, stream>>>(
      query, key, value, wq, wk, wv, bq, bk, bv, Q16, K16, V16);
  attn_kernel<<<dim3(16, 64), 256, 0, stream>>>(Q16, K16, V16, CTX);
  out_proj_kernel<<<dim3(64, 8), 256, 0, stream>>>(CTX, wo, bo, out);
}
